// ArgumentRelationAttention_26414048870462
// MI455X (gfx1250) — matrix operand order repaired and checked by reading, not compiled, not run on hardware
//
#include <hip/hip_runtime.h>
#include <hip/hip_bf16.h>

// ---------------------------------------------------------------------------
// ArgumentRelationAttention for MI455X (gfx1250), fused flash-style kernel.
// All heavy math runs through v_wmma_f32_16x16x32_bf16 (bf16 inputs, f32 acc).
// ---------------------------------------------------------------------------

typedef __attribute__((ext_vector_type(16))) __bf16 v16bf;
typedef __attribute__((ext_vector_type(8)))  __bf16 v8bf;
typedef __attribute__((ext_vector_type(8)))  float  v8f;

#define BB 64
#define NN 512
#define HH 768
#define WAVES 4

// A operand (16x32 bf16): per-lane two 16-byte chunks at K = base and K = base+16.
__device__ __forceinline__ v16bf load_a16(const __bf16* p) {
    v8bf lo = *(const v8bf*)(p);
    v8bf hi = *(const v8bf*)(p + 16);
    return __builtin_shufflevector(lo, hi, 0,1,2,3,4,5,6,7,8,9,10,11,12,13,14,15);
}

// B operand (32x16 bf16): per-lane two 16-byte chunks at K = base and K = base+16.
__device__ __forceinline__ v16bf load_b16(const __bf16* p) {
    v8bf lo = *(const v8bf*)(p);
    v8bf hi = *(const v8bf*)(p + 16);
    return __builtin_shufflevector(lo, hi, 0,1,2,3,4,5,6,7,8,9,10,11,12,13,14,15);
}

__device__ __forceinline__ v8f wmma_bf16(v16bf a, v16bf b, v8f c) {
    return __builtin_amdgcn_wmma_f32_16x16x32_bf16(
        /*neg_a=*/false, a, /*neg_b=*/false, b,
        /*c_mod=*/(short)0, c, /*reuse_a=*/false, /*reuse_b=*/false);
}

// ---------------------------------------------------------------------------
// Prep: tiled fp32->bf16 convert + transpose.
//   src  : fp32 [nb, R, C]
//   dstT : bf16 [nb, C, R]   (dstT[b][c][r] = src[b][r][c])
//   dstS : bf16 [nb, R, C]   straight-converted copy (may be null)
// block (32, 8); grid (C/32, R/32, nb)
// ---------------------------------------------------------------------------
__global__ __launch_bounds__(256)
void prep_transpose_cvt(const float* __restrict__ src,
                        __bf16* __restrict__ dstT,
                        __bf16* __restrict__ dstS,
                        int R, int C) {
    __shared__ float tile[32][33];
    const int b  = blockIdx.z;
    const size_t base = (size_t)b * R * C;
    const int r0 = blockIdx.y * 32, c0 = blockIdx.x * 32;
    const int tx = threadIdx.x, ty = threadIdx.y;
    #pragma unroll
    for (int k = 0; k < 32; k += 8) {
        float v = src[base + (size_t)(r0 + ty + k) * C + (c0 + tx)];
        tile[ty + k][tx] = v;
        if (dstS) dstS[base + (size_t)(r0 + ty + k) * C + (c0 + tx)] = (__bf16)v;
    }
    __syncthreads();
    #pragma unroll
    for (int k = 0; k < 32; k += 8) {
        dstT[base + (size_t)(c0 + ty + k) * R + (r0 + tx)] = (__bf16)tile[tx][ty + k];
    }
}

// ---------------------------------------------------------------------------
// Fused kernel. One workgroup = (batch b, 16-row i-tile).
//   Phase 0: XW_lds[16][768]  = x[b, i0:i0+16] @ W          (WMMA, bf16 out)
//   Phase 1: S_lds[16][512]   = XW @ x[b]^T + bias, diag=0  (WMMA, f32)
//   Phase 2: exact softmax over j; P (bf16) overlays XW_lds
//   Phase 3: out[b, i-tile]   = P @ x[b]                    (WMMA, f32 out)
// ---------------------------------------------------------------------------
__global__ __launch_bounds__(32 * WAVES)
void fused_relation_attn(const __bf16* __restrict__ xb,   // [B, N, H] bf16
                         const __bf16* __restrict__ xt,   // [B, H, N] bf16 (x^T)
                         const __bf16* __restrict__ Wt,   // [H, H] bf16 (W^T)
                         const float*  __restrict__ bias, // [1]
                         float* __restrict__ out) {       // [B, N, H] f32
    __shared__ float  S_lds[16 * NN];          // 32 KB
    __shared__ __bf16 XW_lds[16 * HH];         // 24 KB; reused as P after phase 1
    __shared__ float  red[16 * 8];
    __shared__ float  rowmax[16];
    __shared__ float  rowinv[16];

    const int wg   = blockIdx.x;               // 0 .. B*(N/16)-1
    const int b    = wg >> 5;                  // N/16 == 32
    const int i0   = (wg & 31) * 16;
    const int tid  = threadIdx.x;
    const int wave = tid >> 5;
    const int lane = tid & 31;
    const int lm   = lane & 15;                // row / column within tile
    const int lh   = lane >> 4;                // half-wave select

    const __bf16* xbB = xb + (size_t)b * NN * HH;
    const __bf16* xtB = xt + (size_t)b * HH * NN;
    const float   bia = bias[0];

    // ---- Phase 0: XW tile in LDS ----
    for (int ht = wave; ht < HH / 16; ht += WAVES) {
        v8f acc = {};
        for (int k0 = 0; k0 < HH; k0 += 32) {
            v16bf a = load_a16(xbB + (size_t)(i0 + lm) * HH + k0 + lh * 8);
            v16bf w = load_b16(Wt  + (size_t)(ht * 16 + lm) * HH + k0 + lh * 8);
            acc = wmma_bf16(a, w, acc);
        }
        #pragma unroll
        for (int r = 0; r < 8; ++r) {
            const int M = r + lh * 8;          // C/D layout row
            XW_lds[M * HH + ht * 16 + lm] = (__bf16)acc[r];
        }
    }
    __syncthreads();

    // ---- Phase 1: scores S = XW @ x^T + bias (diagonal forced to 0) ----
    for (int jt = wave; jt < NN / 16; jt += WAVES) {
        v8f acc = {};
        for (int k0 = 0; k0 < HH; k0 += 32) {
            v16bf a  = load_a16(&XW_lds[lm * HH + k0 + lh * 8]);
            v16bf xr = load_b16(xbB + (size_t)(jt * 16 + lm) * HH + k0 + lh * 8);
            acc = wmma_bf16(a, xr, acc);
        }
        #pragma unroll
        for (int r = 0; r < 8; ++r) {
            const int M = r + lh * 8;
            const int j = jt * 16 + lm;
            float v = acc[r] + bia;
            if (i0 + M == j) v = 0.0f;         // reference zeroes the diagonal
            S_lds[M * NN + j] = v;
        }
    }
    __syncthreads();

    // ---- Phase 2: exact softmax over j (8 threads per row) ----
    __bf16* P = XW_lds;                        // overlay: XW no longer needed
    {
        const int row = tid >> 3, seg = tid & 7;
        float m = -3.0e38f;
        for (int j = seg * 64; j < seg * 64 + 64; ++j)
            m = fmaxf(m, S_lds[row * NN + j]);
        red[row * 8 + seg] = m;
    }
    __syncthreads();
    if (tid < 16) {
        float m = red[tid * 8];
        #pragma unroll
        for (int s = 1; s < 8; ++s) m = fmaxf(m, red[tid * 8 + s]);
        rowmax[tid] = m;
    }
    __syncthreads();
    {
        const int row = tid >> 3, seg = tid & 7;
        const float m = rowmax[row];
        float s = 0.0f;
        for (int j = seg * 64; j < seg * 64 + 64; ++j)
            s += __expf(S_lds[row * NN + j] - m);
        red[row * 8 + seg] = s;
    }
    __syncthreads();
    if (tid < 16) {
        float s = 0.0f;
        #pragma unroll
        for (int k = 0; k < 8; ++k) s += red[tid * 8 + k];
        rowinv[tid] = 1.0f / s;
    }
    __syncthreads();
    {
        const int row = tid >> 3, seg = tid & 7;
        const float m = rowmax[row], inv = rowinv[row];
        for (int j = seg * 64; j < seg * 64 + 64; ++j)
            P[row * NN + j] = (__bf16)(__expf(S_lds[row * NN + j] - m) * inv);
    }
    __syncthreads();

    // ---- Phase 3: out = P @ x ----
    float* outB = out + (size_t)b * NN * HH;
    for (int ht = wave; ht < HH / 16; ht += WAVES) {
        v8f acc = {};
        for (int j0 = 0; j0 < NN; j0 += 32) {
            v16bf a  = load_a16(&P[lm * NN + j0 + lh * 8]);
            v16bf xc = load_b16(xtB + (size_t)(ht * 16 + lm) * NN + j0 + lh * 8);
            acc = wmma_bf16(a, xc, acc);
        }
        #pragma unroll
        for (int r = 0; r < 8; ++r) {
            const int M = r + lh * 8;
            outB[(size_t)(i0 + M) * HH + ht * 16 + lm] = acc[r];
        }
    }
}

// ---------------------------------------------------------------------------
extern "C" void kernel_launch(void* const* d_in, const int* in_sizes, int n_in,
                              void* d_out, int out_size, void* d_ws, size_t ws_size,
                              hipStream_t stream) {
    const float* x    = (const float*)d_in[0];   // [B, N, H]
    const float* W    = (const float*)d_in[1];   // [H, H]
    const float* bias = (const float*)d_in[2];   // [1]
    float* out        = (float*)d_out;           // [B, N, H]

    char* ws = (char*)d_ws;
    const size_t xb_bytes = (size_t)BB * NN * HH * sizeof(__bf16);  // 50.3 MB
    __bf16* xb = (__bf16*)(ws);                                     // [B,N,H]
    __bf16* xt = (__bf16*)(ws + xb_bytes);                          // [B,H,N]
    __bf16* Wt = (__bf16*)(ws + 2 * xb_bytes);                      // [H,H]

    dim3 blk(32, 8, 1);
    // x: convert + per-batch transpose
    prep_transpose_cvt<<<dim3(HH / 32, NN / 32, BB), blk, 0, stream>>>(x, xt, xb, NN, HH);
    // W: transpose only (we consume W^T as the K-contiguous B operand)
    prep_transpose_cvt<<<dim3(HH / 32, HH / 32, 1), blk, 0, stream>>>(W, Wt, nullptr, HH, HH);

    fused_relation_attn<<<BB * (NN / 16), 32 * WAVES, 0, stream>>>(xb, xt, Wt, bias, out);
}
